// TransformerEncoder_72670846648591
// MI455X (gfx1250) — matrix operand order repaired and checked by reading, not compiled, not run on hardware
//
#include <hip/hip_runtime.h>
#include <cmath>

typedef __bf16 bf16_t;
typedef __attribute__((ext_vector_type(16))) __bf16 v16bf;
typedef __attribute__((ext_vector_type(8)))  __bf16 v8bf;
typedef __attribute__((ext_vector_type(8)))  float  v8f;

#define DEV __device__ __forceinline__

// ---------------- problem constants ----------------
constexpr int BATCH = 8;
constexpr int HW    = 96;                 // H == W == 96
constexpr int C     = 512;
constexpr int L     = HW * HW;            // 9216
constexpr int MROWS = BATCH * L;          // 73728
constexpr int HEADS = 16;
constexpr int HD    = 32;
constexpr int WSZ   = 8;
constexpr int NTOK  = WSZ * WSZ;          // 64
constexpr int NWIN  = (HW / WSZ) * (HW / WSZ);  // 144
constexpr int BW    = BATCH * NWIN;       // 1152 windows
constexpr float QSCALE = 0.17677669529663687f;  // 32^-0.5

// ---------------- WMMA fragment loaders ----------------
// A-matrix 16x32 (MxK) bf16: lane = m (0..15) in each half-wave;
// lanes 0-15: K = {0..7, 16..23}; lanes 16-31: K = {8..15, 24..31}
DEV v16bf load_frag_a(const bf16_t* base, int stride, int lane) {
  int m  = lane & 15;
  int kb = (lane & 16) >> 1;              // 0 or 8
  const bf16_t* p = base + m * stride + kb;
  v8bf lo = *reinterpret_cast<const v8bf*>(p);
  v8bf hi = *reinterpret_cast<const v8bf*>(p + 16);
  v16bf r;
#pragma unroll
  for (int e = 0; e < 8; ++e) { r[e] = lo[e]; r[e + 8] = hi[e]; }
  return r;
}

// B-matrix 32x16 (KxN) bf16, supplied as B^T (N rows x K cols) row-major:
// lane = n (0..15); lanes 0-15: K = {0..7, 16..23}; lanes 16-31: K = {8..15, 24..31}
DEV v16bf load_frag_b(const bf16_t* baseT, int stride, int lane) {
  int n  = lane & 15;
  int kb = (lane & 16) >> 1;              // 0 or 8
  const bf16_t* p = baseT + n * stride + kb;
  v8bf lo = *reinterpret_cast<const v8bf*>(p);
  v8bf hi = *reinterpret_cast<const v8bf*>(p + 16);
  v16bf r;
#pragma unroll
  for (int e = 0; e < 8; ++e) { r[e] = lo[e]; r[e + 8] = hi[e]; }
  return r;
}

DEV v8f wmma_bf16(v16bf a, v16bf b, v8f c) {
  return __builtin_amdgcn_wmma_f32_16x16x32_bf16(false, a, false, b,
                                                 (short)0, c, false, false);
}

// Async global -> LDS copy of 16 bytes per lane (tracked by ASYNCcnt).
// LDS byte address = low 32 bits of the flat pointer (LDS aperture truncation).
DEV void async_copy16(const bf16_t* gsrc, bf16_t* ldst) {
  unsigned laddr = (unsigned)(size_t)ldst;
  asm volatile("global_load_async_to_lds_b128 %0, %1, off"
               :: "v"(laddr), "v"(gsrc) : "memory");
}

// ---------------- prep kernels ----------------
__global__ void convt_kernel(const float* __restrict__ W, bf16_t* __restrict__ Wt,
                             int K, int N) {
  int i = blockIdx.x * 256 + threadIdx.x;
  if (i >= K * N) return;
  int n = i / K, k = i % K;               // Wt[n][k] = W[k][n]
  Wt[i] = (bf16_t)W[(size_t)k * N + n];
}

__global__ void build_bias_kernel(const float* __restrict__ table,
                                  float* __restrict__ bias) {
  int i = blockIdx.x * 256 + threadIdx.x;
  if (i >= HEADS * NTOK * NTOK) return;
  int h = i >> 12, r = (i >> 6) & 63, c = i & 63;
  int ci = 15 * (r >> 3) + (r & 7);
  int jf = 63 - c;
  int cj = 15 * (jf >> 3) + (jf & 7);
  bias[i] = table[(ci + cj) * HEADS + h];
}

DEV int region96(int v) { return v < 88 ? 0 : (v < 92 ? 1 : 2); }

__global__ void build_mask_kernel(float* __restrict__ mask) {
  int i = blockIdx.x * 256 + threadIdx.x;
  if (i >= NWIN * NTOK * NTOK) return;
  int w = i >> 12, a = (i >> 6) & 63, b = i & 63;
  int wy = w / 12, wx = w % 12;
  int ca = region96(wy * 8 + (a >> 3)) * 3 + region96(wx * 8 + (a & 7));
  int cb = region96(wy * 8 + (b >> 3)) * 3 + region96(wx * 8 + (b & 7));
  mask[i] = (ca != cb) ? -100.0f : 0.0f;
}

// ---------------- layernorm (wave per token) ----------------
// mode 0: read x with +shift, write to window-partitioned xw (bf16)
// mode 1: read src row g, write dst row g (bf16)
__global__ __launch_bounds__(256) void ln_kernel(const float* __restrict__ src,
                                                 bf16_t* __restrict__ dst,
                                                 const float* __restrict__ wgt,
                                                 const float* __restrict__ bia,
                                                 int mode) {
  int wave = threadIdx.x >> 5, lane = threadIdx.x & 31;
  int g = blockIdx.x * 8 + wave;          // token id in destination ordering
  int b = g / L, rem = g % L;
  size_t srow, drow;
  if (mode == 0) {
    int hp = rem / HW, wp = rem % HW;     // shifted-image coords
    srow = (size_t)b * L + ((hp + 4) % HW) * HW + ((wp + 4) % HW);
    int widx = (hp >> 3) * 12 + (wp >> 3);
    drow = (size_t)(b * NWIN + widx) * NTOK + (hp & 7) * 8 + (wp & 7);
  } else {
    srow = g; drow = g;
  }
  const float* xr = src + srow * C;
  float vals[16];
  float s = 0.f;
#pragma unroll
  for (int j = 0; j < 16; ++j) { vals[j] = xr[lane + j * 32]; s += vals[j]; }
#pragma unroll
  for (int o = 16; o > 0; o >>= 1) s += __shfl_xor(s, o, 32);
  float mu = s * (1.0f / 512.0f);
  float vs = 0.f;
#pragma unroll
  for (int j = 0; j < 16; ++j) { float d = vals[j] - mu; vs += d * d; }
#pragma unroll
  for (int o = 16; o > 0; o >>= 1) vs += __shfl_xor(vs, o, 32);
  float inv = rsqrtf(vs * (1.0f / 512.0f) + 1e-5f);
  bf16_t* dr = dst + drow * C;
#pragma unroll
  for (int j = 0; j < 16; ++j) {
    int c = lane + j * 32;
    dr[c] = (bf16_t)((vals[j] - mu) * inv * wgt[c] + bia[c]);
  }
}

// ---------------- tiled bf16 GEMM with epilogues ----------------
// C[M,N] = A[M,K] @ Bt[N,K]^T ; block tile 128x128, 8 waves, K-step 32.
// Double-buffered LDS fed by GLOBAL_LOAD_ASYNC_TO_LDS_B128 (ASYNCcnt pipelined).
// EPI 0: qkv scatter   1: proj + reverse/unshift + residual -> f32
// EPI 2: gelu -> bf16  3: + residual -> f32
constexpr int KP = 40;                    // padded LDS K stride (halves)

template <int EPI>
__global__ __launch_bounds__(256) void gemm_bf16_kernel(
    const bf16_t* __restrict__ A, const bf16_t* __restrict__ Bt,
    int M, int N, int K, const float* __restrict__ bias,
    float* __restrict__ outF, bf16_t* __restrict__ outH,
    bf16_t* __restrict__ qOut, bf16_t* __restrict__ kOut,
    bf16_t* __restrict__ vOut, const float* __restrict__ resid) {
  __shared__ bf16_t lA[2][128 * KP];
  __shared__ bf16_t lB[2][128 * KP];
  const int tid = threadIdx.x, lane = tid & 31, wave = tid >> 5;
  const int wm = wave & 3, wn = wave >> 2;
  const int m0 = blockIdx.y * 128, n0 = blockIdx.x * 128;

  v8f acc[2][4];
#pragma unroll
  for (int a = 0; a < 2; ++a)
#pragma unroll
    for (int b = 0; b < 4; ++b)
#pragma unroll
      for (int e = 0; e < 8; ++e) acc[a][b][e] = 0.f;

  // issue one stage of async global->LDS copies (4 instructions per thread)
  auto stage = [&](int ks, int buf) {
#pragma unroll
    for (int idx = tid; idx < 128 * 4; idx += 256) {
      int r = idx >> 2, sg = idx & 3;
      async_copy16(&A[(size_t)(m0 + r) * K + ks + sg * 8],
                   &lA[buf][r * KP + sg * 8]);
      async_copy16(&Bt[(size_t)(n0 + r) * K + ks + sg * 8],
                   &lB[buf][r * KP + sg * 8]);
    }
  };

  stage(0, 0);
  int it = 0;
  for (int ks = 0; ks < K; ks += 32, ++it) {
    const int cur = it & 1;
    if (ks + 32 < K) {
      stage(ks + 32, cur ^ 1);                         // prefetch next stage
      asm volatile("s_wait_asynccnt 0x4" ::: "memory"); // current stage resident
    } else {
      asm volatile("s_wait_asynccnt 0x0" ::: "memory");
    }
    __syncthreads();                                    // publish all waves' LDS
    v16bf af[2], bfr[4];
#pragma unroll
    for (int mt = 0; mt < 2; ++mt)
      af[mt] = load_frag_a(&lA[cur][(wm * 32 + mt * 16) * KP], KP, lane);
#pragma unroll
    for (int nt = 0; nt < 4; ++nt)
      bfr[nt] = load_frag_b(&lB[cur][(wn * 64 + nt * 16) * KP], KP, lane);
#pragma unroll
    for (int mt = 0; mt < 2; ++mt)
#pragma unroll
      for (int nt = 0; nt < 4; ++nt)
        acc[mt][nt] = wmma_bf16(af[mt], bfr[nt], acc[mt][nt]);
    __syncthreads();                                    // protect buffer reuse
  }

  const int iOff = (lane >> 4) << 3, jl = lane & 15;
#pragma unroll
  for (int mt = 0; mt < 2; ++mt)
#pragma unroll
    for (int nt = 0; nt < 4; ++nt)
#pragma unroll
      for (int r = 0; r < 8; ++r) {
        int row = m0 + wm * 32 + mt * 16 + r + iOff;
        int col = n0 + wn * 64 + nt * 16 + jl;
        float val = acc[mt][nt][r] + bias[col];
        if constexpr (EPI == 0) {
          int w = row >> 6, t = row & 63;
          int c5 = col & 511, h = c5 >> 5, d = c5 & 31;
          size_t hb = (size_t)(w * HEADS + h);
          if (col < 512)
            qOut[hb * 2048 + t * 32 + d] = (bf16_t)(val * QSCALE);
          else if (col < 1024)
            kOut[hb * 2048 + t * 32 + d] = (bf16_t)val;
          else
            vOut[hb * 2048 + d * 64 + t] = (bf16_t)val;   // transposed: [hd][tok]
        } else if constexpr (EPI == 1) {
          int w = row >> 6, t = row & 63;
          int b = w / NWIN, widx = w % NWIN;
          int hp = (widx / 12) * 8 + (t >> 3), wp = (widx % 12) * 8 + (t & 7);
          int hh = (hp + 4) % HW, ww = (wp + 4) % HW;
          size_t o = ((size_t)b * L + hh * HW + ww) * C + col;
          outF[o] = resid[o] + val;
        } else if constexpr (EPI == 2) {
          float g = 0.5f * val * (1.0f + erff(val * 0.70710678118654752f));
          outH[(size_t)row * C + col] = (bf16_t)g;
        } else {
          size_t o = (size_t)row * C + col;
          outF[o] = resid[o] + val;
        }
      }
}

// ---------------- fused window attention ----------------
// 1 block per window (1152), 8 waves, 2 heads per wave.
// S = Q@K^T (+bias+mask, NO softmax), O = S@V; S round-trips LDS as bf16 A-frag.
constexpr int SP = 72;                    // padded S-tile stride (halves)

__global__ __launch_bounds__(256) void attn_kernel(
    const bf16_t* __restrict__ q, const bf16_t* __restrict__ k,
    const bf16_t* __restrict__ v, const float* __restrict__ bias,
    const float* __restrict__ mask, bf16_t* __restrict__ out) {
  int w = blockIdx.x;
  int wave = threadIdx.x >> 5, lane = threadIdx.x & 31;
  __shared__ bf16_t sS[8][16 * SP];
  const int mIdx = w % NWIN;
  const int iOff = (lane >> 4) << 3, jl = lane & 15;

  for (int hh = 0; hh < 2; ++hh) {
    int h = wave * 2 + hh;
    const bf16_t* qh = q + (size_t)(w * HEADS + h) * 2048;
    const bf16_t* kh = k + (size_t)(w * HEADS + h) * 2048;
    const bf16_t* vh = v + (size_t)(w * HEADS + h) * 2048;

    v16bf kf[4];
#pragma unroll
    for (int nt = 0; nt < 4; ++nt)
      kf[nt] = load_frag_b(kh + nt * 16 * HD, HD, lane);    // B^T rows = tokens
    v16bf vf[2][2];
#pragma unroll
    for (int nt = 0; nt < 2; ++nt)
#pragma unroll
      for (int kk = 0; kk < 2; ++kk)
        vf[nt][kk] = load_frag_b(vh + (nt * 16) * NTOK + kk * 32, NTOK, lane);

#pragma unroll
    for (int mt = 0; mt < 4; ++mt) {
      v16bf qf = load_frag_a(qh + mt * 16 * HD, HD, lane);
#pragma unroll
      for (int nt = 0; nt < 4; ++nt) {
        v8f s;
#pragma unroll
        for (int e = 0; e < 8; ++e) s[e] = 0.f;
        s = wmma_bf16(qf, kf[nt], s);
#pragma unroll
        for (int r = 0; r < 8; ++r) {
          int iLoc = r + iOff;
          int i = mt * 16 + iLoc;
          int j = nt * 16 + jl;
          float val = s[r] + bias[(size_t)h * 4096 + i * 64 + j] +
                      mask[(size_t)mIdx * 4096 + i * 64 + j];
          sS[wave][iLoc * SP + j] = (bf16_t)val;
        }
      }
      asm volatile("s_wait_dscnt 0" ::: "memory");
      v16bf sf0 = load_frag_a(&sS[wave][0], SP, lane);      // K = tokens 0..31
      v16bf sf1 = load_frag_a(&sS[wave][32], SP, lane);     // K = tokens 32..63
#pragma unroll
      for (int nto = 0; nto < 2; ++nto) {
        v8f o;
#pragma unroll
        for (int e = 0; e < 8; ++e) o[e] = 0.f;
        o = wmma_bf16(sf0, vf[nto][0], o);
        o = wmma_bf16(sf1, vf[nto][1], o);
#pragma unroll
        for (int r = 0; r < 8; ++r) {
          size_t row = (size_t)w * NTOK + mt * 16 + r + iOff;
          int col = h * HD + nto * 16 + jl;
          out[row * C + col] = (bf16_t)o[r];
        }
      }
    }
  }
}

// ---------------- host ----------------
static inline char* carve(char*& p, size_t bytes) {
  char* r = p;
  p += (bytes + 255) & ~(size_t)255;
  return r;
}

extern "C" void kernel_launch(void* const* d_in, const int* in_sizes, int n_in,
                              void* d_out, int out_size, void* d_ws, size_t ws_size,
                              hipStream_t stream) {
  const float* x       = (const float*)d_in[0];
  const float* norm_w  = (const float*)d_in[1];
  const float* norm_b  = (const float*)d_in[2];
  const float* qkv_w   = (const float*)d_in[3];
  const float* qkv_b   = (const float*)d_in[4];
  const float* proj_w  = (const float*)d_in[5];
  const float* proj_b  = (const float*)d_in[6];
  const float* rel_tab = (const float*)d_in[7];
  const float* fc1_w   = (const float*)d_in[8];
  const float* fc1_b   = (const float*)d_in[9];
  const float* fc2_w   = (const float*)d_in[10];
  const float* fc2_b   = (const float*)d_in[11];
  (void)in_sizes; (void)n_in; (void)out_size; (void)ws_size;

  char* p = (char*)d_ws;
  bf16_t* qkv_wt  = (bf16_t*)carve(p, (size_t)3 * C * C * 2);
  bf16_t* proj_wt = (bf16_t*)carve(p, (size_t)C * C * 2);
  bf16_t* fc1_wt  = (bf16_t*)carve(p, (size_t)C * C * 2);
  bf16_t* fc2_wt  = (bf16_t*)carve(p, (size_t)C * C * 2);
  float*  biasA   = (float*)carve(p, (size_t)HEADS * 4096 * 4);
  float*  maskA   = (float*)carve(p, (size_t)NWIN * 4096 * 4);
  const size_t actH = (size_t)MROWS * C * 2;                 // bf16 activation
  bf16_t* bufA = (bf16_t*)carve(p, actH);                    // xw / attn_out / h2
  char*   bufB = carve(p, actH * 3);                         // q|k|v, later x2|m1
  bf16_t* qB = (bf16_t*)bufB;
  bf16_t* kB = (bf16_t*)(bufB + actH);
  bf16_t* vB = (bf16_t*)(bufB + 2 * actH);
  float*  x2 = (float*)bufB;                                 // 2*actH bytes (q+k dead)
  bf16_t* m1 = vB;                                           // v dead by then

  // weights: convert + transpose to (N x K) bf16
  convt_kernel<<<(C * 3 * C + 255) / 256, 256, 0, stream>>>(qkv_w, qkv_wt, C, 3 * C);
  convt_kernel<<<(C * C + 255) / 256, 256, 0, stream>>>(proj_w, proj_wt, C, C);
  convt_kernel<<<(C * C + 255) / 256, 256, 0, stream>>>(fc1_w, fc1_wt, C, C);
  convt_kernel<<<(C * C + 255) / 256, 256, 0, stream>>>(fc2_w, fc2_wt, C, C);
  build_bias_kernel<<<(HEADS * 4096 + 255) / 256, 256, 0, stream>>>(rel_tab, biasA);
  build_mask_kernel<<<(NWIN * 4096 + 255) / 256, 256, 0, stream>>>(maskA);

  // LN1 + shift + window partition -> xw
  ln_kernel<<<MROWS / 8, 256, 0, stream>>>(x, bufA, norm_w, norm_b, 0);

  // QKV projection
  gemm_bf16_kernel<0><<<dim3((3 * C) / 128, MROWS / 128), 256, 0, stream>>>(
      bufA, qkv_wt, MROWS, 3 * C, C, qkv_b, nullptr, nullptr, qB, kB, vB, nullptr);

  // fused window attention -> bufA (attn_out, token-major 73728x512 bf16)
  attn_kernel<<<BW, 256, 0, stream>>>(qB, kB, vB, biasA, maskA, bufA);

  // proj + window reverse + unshift + residual -> x2 (fp32)
  gemm_bf16_kernel<1><<<dim3(C / 128, MROWS / 128), 256, 0, stream>>>(
      bufA, proj_wt, MROWS, C, C, proj_b, x2, nullptr, nullptr, nullptr, nullptr, x);

  // LN2 -> h2 (bufA)
  ln_kernel<<<MROWS / 8, 256, 0, stream>>>(x2, bufA, norm_w, norm_b, 1);

  // fc1 + gelu -> m1
  gemm_bf16_kernel<2><<<dim3(C / 128, MROWS / 128), 256, 0, stream>>>(
      bufA, fc1_wt, MROWS, C, C, fc1_b, nullptr, m1, nullptr, nullptr, nullptr, nullptr);

  // fc2 + residual -> d_out
  gemm_bf16_kernel<3><<<dim3(C / 128, MROWS / 128), 256, 0, stream>>>(
      m1, fc2_wt, MROWS, C, C, fc2_b, (float*)d_out, nullptr, nullptr, nullptr, nullptr, x2);
}
